// SelfAttention_8839042695220
// MI455X (gfx1250) — matrix operand order repaired and checked by reading, not compiled, not run on hardware
//
#include <hip/hip_runtime.h>

// ---------------------------------------------------------------------------
// CDNA5 (gfx1250) bf16 WMMA self-attention:
//   qkv = x @ W_attn + b ; flash attention ; out = y @ W_proj + b
// wave32, v_wmma_f32_16x16x32_bf16, L2-resident operand streaming.
// ---------------------------------------------------------------------------

typedef __bf16 v16bf __attribute__((ext_vector_type(16)));
typedef float  v8f   __attribute__((ext_vector_type(8)));

union FragU { uint4 q[2]; v16bf v; };

// A-matrix 16x32 bf16 fragment (row-major source, leading dim = ld elements).
// ISA layout: lane m = lane%16, half = lane/16; VGPR v holds
// K = (v/4)*16 + half*8 + (v%4)*2  and K+1  -> two contiguous 16B loads.
__device__ __forceinline__ v16bf load_frag_a(const __bf16* p, int ld) {
  const int lane = threadIdx.x & 31;
  const int m = lane & 15, half = lane >> 4;
  const __bf16* row = p + (size_t)m * ld + half * 8;
  FragU f;
  f.q[0] = *(const uint4*)(row);        // K rel 0..7   (pairs 0,2,4,6)
  f.q[1] = *(const uint4*)(row + 16);   // K rel 16..23
  return f.v;
}

// B-matrix 32x16 bf16 fragment from a K-contiguous layout Bt[n][k] (ld = K dim).
// ISA layout: lane n = lane%16, half = lane/16; same K order as load_frag_a.
__device__ __forceinline__ v16bf load_frag_b(const __bf16* p, int ld) {
  const int lane = threadIdx.x & 31;
  const int n = lane & 15, half = lane >> 4;
  const __bf16* row = p + (size_t)n * ld + half * 8;
  FragU f;
  f.q[0] = *(const uint4*)(row);        // K = half*8 + 0..7
  f.q[1] = *(const uint4*)(row + 16);   // K = 16 + half*8 + 0..7
  return f.v;
}

__device__ __forceinline__ v8f wmma_bf16(v16bf a, v16bf b, v8f c) {
  return __builtin_amdgcn_wmma_f32_16x16x32_bf16(
      /*neg_a=*/false, a, /*neg_b=*/false, b,
      /*c_mod=*/(short)0, c, /*reuse_a=*/false, /*reuse_b=*/false);
}

// ---------------------------------------------------------------------------
// fp32 -> bf16 convert (elementwise)
// ---------------------------------------------------------------------------
__global__ void k_cvt(const float* __restrict__ s, unsigned short* __restrict__ d, int n) {
  __bf16* db = reinterpret_cast<__bf16*>(d);
  for (int i = blockIdx.x * blockDim.x + threadIdx.x; i < n; i += gridDim.x * blockDim.x)
    db[i] = (__bf16)s[i];
}

// fp32 [K][N] -> bf16 [N][K] (transpose-convert so GEMM B-fragments are K-contiguous)
__global__ void k_cvt_t(const float* __restrict__ s, unsigned short* __restrict__ d,
                        int K, int N) {
  __bf16* db = reinterpret_cast<__bf16*>(d);
  int total = K * N;
  for (int i = blockIdx.x * blockDim.x + threadIdx.x; i < total; i += gridDim.x * blockDim.x) {
    int k = i / N, n = i - k * N;
    db[(size_t)n * K + k] = (__bf16)s[i];
  }
}

// ---------------------------------------------------------------------------
// GEMM1: qkv = x(bf16)[8192,1024] @ W_attn_T(bf16)[3072,1024]^T + b_attn
// Scatters q/k into [bh][t][64] and v into transposed [bh][64][t] bf16 buffers.
// Block = 128 threads (4 waves, 2x2), block tile 128x128, wave tile 64x64.
// ---------------------------------------------------------------------------
__global__ void __launch_bounds__(128)
k_gemm_qkv(const unsigned short* __restrict__ Xu, const unsigned short* __restrict__ Wu,
           const float* __restrict__ bias, unsigned short* __restrict__ Qu,
           unsigned short* __restrict__ Ku, unsigned short* __restrict__ Vtu) {
  const __bf16* X = reinterpret_cast<const __bf16*>(Xu);
  const __bf16* W = reinterpret_cast<const __bf16*>(Wu);
  __bf16* Qb = reinterpret_cast<__bf16*>(Qu);
  __bf16* Kb = reinterpret_cast<__bf16*>(Ku);
  __bf16* Vb = reinterpret_cast<__bf16*>(Vtu);
  const int KD = 1024;
  const int lane = threadIdx.x & 31, wave = threadIdx.x >> 5;
  const int m0 = blockIdx.x * 128 + (wave >> 1) * 64;
  const int n0 = blockIdx.y * 128 + (wave & 1) * 64;

  v8f acc[4][4] = {};
  for (int k0 = 0; k0 < KD; k0 += 32) {
    v16bf a[4], b[4];
#pragma unroll
    for (int mi = 0; mi < 4; ++mi)
      a[mi] = load_frag_a(X + (size_t)(m0 + mi * 16) * KD + k0, KD);
#pragma unroll
    for (int ni = 0; ni < 4; ++ni)
      b[ni] = load_frag_b(W + (size_t)(n0 + ni * 16) * KD + k0, KD);
#pragma unroll
    for (int mi = 0; mi < 4; ++mi)
#pragma unroll
      for (int ni = 0; ni < 4; ++ni)
        acc[mi][ni] = wmma_bf16(a[mi], b[ni], acc[mi][ni]);
  }

  const int nl = lane & 15, half = lane >> 4;
#pragma unroll
  for (int mi = 0; mi < 4; ++mi) {
#pragma unroll
    for (int ni = 0; ni < 4; ++ni) {
      const int n = n0 + ni * 16 + nl;
      const int part = n >> 10, c = n & 1023, h = c >> 6, dd = c & 63;
      const float bv = bias[n];
#pragma unroll
      for (int r = 0; r < 8; ++r) {
        const int m = m0 + mi * 16 + r + 8 * half;   // global row = b*2048 + t
        const int bb = m >> 11, t = m & 2047;
        const int bh = bb * 16 + h;
        const __bf16 val = (__bf16)(acc[mi][ni][r] + bv);
        if (part == 0)      Qb[((size_t)bh * 2048 + t) * 64 + dd] = val;
        else if (part == 1) Kb[((size_t)bh * 2048 + t) * 64 + dd] = val;
        else                Vb[((size_t)bh * 64 + dd) * 2048 + t] = val; // V transposed
      }
    }
  }
}

// ---------------------------------------------------------------------------
// Flash attention: grid = (B*H, T/128), block = 256 (8 waves, 16 queries/wave).
// S = Q K^T via WMMA, online softmax (exp2, xor-shuffle row reductions),
// P re-laid out through wave-private LDS, Y += P V via WMMA.
// K/V stream from the 192MB L2 (entire bf16 QKV = 48MB is L2-resident).
// ---------------------------------------------------------------------------
__global__ void __launch_bounds__(256)
k_flash(const unsigned short* __restrict__ Qu, const unsigned short* __restrict__ Ku,
        const unsigned short* __restrict__ Vtu, unsigned short* __restrict__ Yu) {
  const int T = 2048, D = 64;
  const __bf16* Q  = reinterpret_cast<const __bf16*>(Qu);
  const __bf16* Kc = reinterpret_cast<const __bf16*>(Ku);
  const __bf16* Vt = reinterpret_cast<const __bf16*>(Vtu);
  __bf16* Y = reinterpret_cast<__bf16*>(Yu);

  const int lane = threadIdx.x & 31, wave = threadIdx.x >> 5;
  const int nl = lane & 15, half = lane >> 4;
  const int bh = blockIdx.x;                         // b*16 + h
  const int q0 = blockIdx.y * 128 + wave * 16;

  const __bf16* Qp = Q  + ((size_t)bh * T + q0) * D;
  const __bf16* Kp = Kc + (size_t)bh * T * D;        // [t][d]
  const __bf16* Vp = Vt + (size_t)bh * D * T;        // [d][t] (transposed)

  __shared__ __align__(16) __bf16 psh[8][16 * 32];   // per-wave P scratch (1KB)
  __bf16* ps = &psh[wave][0];

  // Q tile 16x64 held in registers for the whole key sweep (2 A-fragments)
  const v16bf qa0 = load_frag_a(Qp, D);
  const v16bf qa1 = load_frag_a(Qp + 32, D);

  v8f yacc[4] = {};
  float rm[8], rs[8];
#pragma unroll
  for (int r = 0; r < 8; ++r) { rm[r] = -3.0e38f; rs[r] = 0.0f; }
  const float ks = 0.125f * 1.44269504088896f;       // 1/sqrt(64) * log2(e)

  for (int t0 = 0; t0 < T; t0 += 32) {
    if (t0 + 32 < T) {
      __builtin_prefetch(Kp + (size_t)(t0 + 32) * D, 0, 1);  // global_prefetch_b8
      __builtin_prefetch(Vp + (t0 + 32), 0, 1);
    }
    // ---- S[16q x 32keys] = Q(16x64) * K^T, contraction over d in 2 slices ----
    v8f s0 = {}, s1 = {};
    {
      v16bf b00 = load_frag_b(Kp + (size_t)t0 * D, D);
      v16bf b01 = load_frag_b(Kp + (size_t)(t0 + 16) * D, D);
      s0 = wmma_bf16(qa0, b00, s0);
      s1 = wmma_bf16(qa0, b01, s1);
      v16bf b10 = load_frag_b(Kp + (size_t)t0 * D + 32, D);
      v16bf b11 = load_frag_b(Kp + (size_t)(t0 + 16) * D + 32, D);
      s0 = wmma_bf16(qa1, b10, s0);
      s1 = wmma_bf16(qa1, b11, s1);
    }
    // ---- online softmax: C-layout row m = r + 8*half, 16-lane group reductions
    float corr[8];
#pragma unroll
    for (int r = 0; r < 8; ++r) {
      float v = fmaxf(s0[r], s1[r]);
      v = fmaxf(v, __shfl_xor(v, 1, 32));
      v = fmaxf(v, __shfl_xor(v, 2, 32));
      v = fmaxf(v, __shfl_xor(v, 4, 32));
      v = fmaxf(v, __shfl_xor(v, 8, 32));
      const float nm = fmaxf(rm[r], v);
      corr[r] = exp2f((rm[r] - nm) * ks);
      rm[r] = nm;
      const float p0 = exp2f((s0[r] - nm) * ks);
      const float p1 = exp2f((s1[r] - nm) * ks);
      float pp = p0 + p1;
      pp += __shfl_xor(pp, 1, 32);
      pp += __shfl_xor(pp, 2, 32);
      pp += __shfl_xor(pp, 4, 32);
      pp += __shfl_xor(pp, 8, 32);
      rs[r] = rs[r] * corr[r] + pp;
      const int m = r + 8 * half;                    // C-layout -> (m, n) scatter
      ps[m * 32 + nl]      = (__bf16)p0;             // keys t0 + 0..15
      ps[m * 32 + 16 + nl] = (__bf16)p1;             // keys t0 + 16..31
    }
#pragma unroll
    for (int dt = 0; dt < 4; ++dt)
#pragma unroll
      for (int r = 0; r < 8; ++r) yacc[dt][r] *= corr[r];

    // ---- Y(16x64) += P(16x32) * V(32x64); P re-read from LDS in A layout ----
    const v16bf pa = load_frag_a(ps, 32);
#pragma unroll
    for (int dt = 0; dt < 4; ++dt) {
      v16bf vb = load_frag_b(Vp + (size_t)(dt * 16) * T + t0, T);
      yacc[dt] = wmma_bf16(pa, vb, yacc[dt]);
    }
  }

  const int bq = bh >> 4, h = bh & 15;
#pragma unroll
  for (int dt = 0; dt < 4; ++dt)
#pragma unroll
    for (int r = 0; r < 8; ++r) {
      const int t = q0 + r + 8 * half;
      const float val = yacc[dt][r] / rs[r];
      Y[((size_t)(bq * 2048 + t)) * 1024 + h * 64 + dt * 16 + nl] = (__bf16)val;
    }
}

// ---------------------------------------------------------------------------
// GEMM2: out = y(bf16)[8192,1024] @ W_proj_T(bf16)[1024,1024]^T + b_proj (fp32 out)
// ---------------------------------------------------------------------------
__global__ void __launch_bounds__(128)
k_gemm_proj(const unsigned short* __restrict__ Yu, const unsigned short* __restrict__ Wu,
            const float* __restrict__ bias, float* __restrict__ out) {
  const __bf16* Yb = reinterpret_cast<const __bf16*>(Yu);
  const __bf16* W  = reinterpret_cast<const __bf16*>(Wu);
  const int KD = 1024;
  const int lane = threadIdx.x & 31, wave = threadIdx.x >> 5;
  const int m0 = blockIdx.x * 128 + (wave >> 1) * 64;
  const int n0 = blockIdx.y * 128 + (wave & 1) * 64;

  v8f acc[4][4] = {};
  for (int k0 = 0; k0 < KD; k0 += 32) {
    v16bf a[4], b[4];
#pragma unroll
    for (int mi = 0; mi < 4; ++mi)
      a[mi] = load_frag_a(Yb + (size_t)(m0 + mi * 16) * KD + k0, KD);
#pragma unroll
    for (int ni = 0; ni < 4; ++ni)
      b[ni] = load_frag_b(W + (size_t)(n0 + ni * 16) * KD + k0, KD);
#pragma unroll
    for (int mi = 0; mi < 4; ++mi)
#pragma unroll
      for (int ni = 0; ni < 4; ++ni)
        acc[mi][ni] = wmma_bf16(a[mi], b[ni], acc[mi][ni]);
  }

  const int nl = lane & 15, half = lane >> 4;
#pragma unroll
  for (int mi = 0; mi < 4; ++mi)
#pragma unroll
    for (int ni = 0; ni < 4; ++ni) {
      const int n = n0 + ni * 16 + nl;
      const float bv = bias[n];
#pragma unroll
      for (int r = 0; r < 8; ++r) {
        const int m = m0 + mi * 16 + r + 8 * half;
        out[(size_t)m * 1024 + n] = acc[mi][ni][r] + bv;
      }
    }
}

// ---------------------------------------------------------------------------
// Host launcher
// ---------------------------------------------------------------------------
extern "C" void kernel_launch(void* const* d_in, const int* in_sizes, int n_in,
                              void* d_out, int out_size, void* d_ws, size_t ws_size,
                              hipStream_t stream) {
  (void)in_sizes; (void)n_in; (void)out_size; (void)ws_size;
  const float* x      = (const float*)d_in[0];   // [4,2048,1024]
  const float* W_attn = (const float*)d_in[1];   // [1024,3072]
  const float* b_attn = (const float*)d_in[2];   // [3072]
  const float* W_proj = (const float*)d_in[3];   // [1024,1024]
  const float* b_proj = (const float*)d_in[4];   // [1024]
  float* out = (float*)d_out;                    // [4,2048,1024] fp32

  const size_t MT = 8192;                        // B*T rows
  size_t off = 0;
  auto carve = [&](size_t bytes) -> unsigned short* {
    unsigned short* p = (unsigned short*)((char*)d_ws + off);
    off += (bytes + 255) & ~(size_t)255;
    return p;
  };
  unsigned short* xb     = carve(MT * 1024 * 2);            // x bf16
  unsigned short* WattnT = carve((size_t)3072 * 1024 * 2);  // W_attn^T bf16
  unsigned short* WprojT = carve((size_t)1024 * 1024 * 2);  // W_proj^T bf16
  unsigned short* Qb     = carve((size_t)64 * 2048 * 64 * 2);
  unsigned short* Kb     = carve((size_t)64 * 2048 * 64 * 2);
  unsigned short* Vt     = carve((size_t)64 * 64 * 2048 * 2); // V transposed [bh][d][t]
  unsigned short* Yb     = carve(MT * 1024 * 2);            // attention out bf16

  // 1) precision conversion (HBM-trivial: <40MB at 23.3 TB/s)
  {
    int n = (int)(MT * 1024);
    k_cvt<<<dim3((n + 255) / 256), dim3(256), 0, stream>>>(x, xb, n);
  }
  {
    int n = 1024 * 3072;
    k_cvt_t<<<dim3((n + 255) / 256), dim3(256), 0, stream>>>(W_attn, WattnT, 1024, 3072);
  }
  {
    int n = 1024 * 1024;
    k_cvt_t<<<dim3((n + 255) / 256), dim3(256), 0, stream>>>(W_proj, WprojT, 1024, 1024);
  }
  // 2) fused QKV projection (WMMA) with per-head scatter
  k_gemm_qkv<<<dim3(64, 24), dim3(128), 0, stream>>>(xb, WattnT, b_attn, Qb, Kb, Vt);
  // 3) flash attention (WMMA + exp2 + LDS P-transpose)
  k_flash<<<dim3(64, 16), dim3(256), 0, stream>>>(Qb, Kb, Vt, Yb);
  // 4) output projection (WMMA, fp32 epilogue)
  k_gemm_proj<<<dim3(64, 8), dim3(128), 0, stream>>>(Yb, WprojT, b_proj, out);
}
